// newLocalGrouper_57629871178073
// MI455X (gfx1250) — matrix operand order repaired and checked by reading, not compiled, not run on hardware
//
#include <hip/hip_runtime.h>
#include <hip/hip_bf16.h>

// ---------------------------------------------------------------------------
// Sizes from the reference
// ---------------------------------------------------------------------------
#define B_    8
#define NPTS  4096
#define CH    64
#define NGRP  512
#define BS    (B_ * NGRP)          // 4096 query points
#define CATD  384                  // 128 + 256
#define KOUT  32                   // NSAMPLE[-1]

typedef _Float16 h16;
typedef __attribute__((ext_vector_type(16))) _Float16 v16h;
typedef __attribute__((ext_vector_type(8)))  _Float16 v8h;
typedef __attribute__((ext_vector_type(8)))  float    v8f;
typedef unsigned int u32;
typedef __attribute__((ext_vector_type(4))) u32 v4u;
typedef __attribute__((ext_vector_type(8))) u32 v8u;

union Frag16 { v16h v; v8h h[2]; };

static inline unsigned cdiv(unsigned a, unsigned b) { return (a + b - 1) / b; }

// ---------------------------------------------------------------------------
// Generic WMMA GEMM:  out[M,Nc] = A[M,Kp](f16) * W[Nc,Kp]^T(f16) (+bias)(+relu)
//
// The whole weight matrix W [Nc,Kp] (<=128KB f16, fits in the 320KB WGP LDS)
// is staged into LDS once per workgroup by wave 0 using the Tensor Data Mover
// (tensor_load_to_lds, D# built per CDNA5 ISA), completion via
// s_wait_tensorcnt + workgroup barrier. B-fragments then come from ds_load,
// A-fragments stream from global with prefetch.
//
// One wave computes a 16x64 output tile; K-loop in steps of 32 using
// v_wmma_f32_16x16x32_f16. Fragment lane layouts follow the CDNA5 ISA tables:
//   A (16x32 f16): lanes 0-15 hold M, halves K[0..7]/K[16..23]; lanes 16-31
//                  hold K[8..15]/K[24..31].
//   B (32x16 f16): lane = N column, lanes 0-15 hold K[0..7]/K[16..23], 16-31 K[8..15]/K[24..31].
//   C/D (16x16 f32): VGPR i -> row (i + 8*(lane>=16)), col = lane&15.
// ---------------------------------------------------------------------------
extern __shared__ h16 smW[];

__launch_bounds__(128)
__global__ void gemm_wmma(const h16* __restrict__ A, const h16* __restrict__ W,
                          const float* __restrict__ bias,
                          float* __restrict__ outF, h16* __restrict__ outH,
                          int M, int Nc, int Kp, int relu)
{
    const int lane = threadIdx.x & 31;
    const int wave = threadIdx.x >> 5;

    // ---- TDM: stage W[Nc,Kp] f16 -> LDS (wave 0 issues; all waves barrier) ----
    if (wave == 0) {
        union { v4u v; u32 u[4]; } g0 = {};
        union { v8u v; u32 u[8]; } g1 = {};
        const unsigned long long ga = (unsigned long long)(uintptr_t)W;
        const u32 la = (u32)(uintptr_t)smW;      // flat addr low 32 = LDS byte offset
        const u32 kp = (u32)Kp, nc = (u32)Nc;
        // D# group 0: count=1 | lds_addr | global_addr[56:0] | type=2
        g0.u[0] = 1u;
        g0.u[1] = la;
        g0.u[2] = (u32)ga;
        g0.u[3] = ((u32)(ga >> 32) & 0x01FFFFFFu) | 0x80000000u;
        // D# group 1: data_size=2B; tensor_dim0=Kp; tensor_dim1=Nc;
        //             tile_dim0=Kp; tile_dim1=Nc; tensor_dim0_stride=Kp
        g1.u[0] = 0x00010000u;                              // data_size code 1 (2B)
        g1.u[1] = (kp & 0xFFFFu) << 16;                     // tensor_dim0[15:0]
        g1.u[2] = ((kp >> 16) & 0xFFFFu) | ((nc & 0xFFFFu) << 16); // dim0 hi | dim1 lo
        g1.u[3] = ((nc >> 16) & 0xFFFFu) | ((kp & 0xFFFFu) << 16); // dim1 hi | tile_dim0
        g1.u[4] = nc & 0xFFFFu;                             // tile_dim1 (tile_dim2=0)
        g1.u[5] = kp;                                       // tensor_dim0_stride[31:0]
        g1.u[6] = 0u;                                       // stride hi | dim1_stride lo
        g1.u[7] = 0u;
        asm volatile("tensor_load_to_lds %0, %1" :: "s"(g0.v), "s"(g1.v) : "memory");
        __builtin_amdgcn_s_wait_tensorcnt(0);
    }
    __syncthreads();

    const int tm = (blockIdx.x * 4 + wave) << 4;
    const int tn = blockIdx.y << 6;
    if (tm >= M) return;
    const int l15 = lane & 15;
    const int hi  = lane >> 4;            // 0 or 1
    const h16* arow = A + (size_t)(tm + l15) * Kp;
    const int klo = hi * 8;               // A half-frag K offset
    const int kh  = hi * 8;               // B half-frag K offset
    v8f acc[4] = {};
    for (int kk = 0; kk < Kp; kk += 32) {
        Frag16 af;
        af.h[0] = *(const v8h*)(arow + kk + klo);
        af.h[1] = *(const v8h*)(arow + kk + 16 + klo);
        if (kk + 32 < Kp) __builtin_prefetch(arow + kk + 32, 0, 1);
        Frag16 bf[4];
#pragma unroll
        for (int j = 0; j < 4; ++j) {
            const h16* wr = smW + (size_t)(tn + j * 16 + l15) * Kp + kk + kh;
            bf[j].h[0] = *(const v8h*)(wr);
            bf[j].h[1] = *(const v8h*)(wr + 16);
        }
#pragma unroll
        for (int j = 0; j < 4; ++j) {
            acc[j] = __builtin_amdgcn_wmma_f32_16x16x32_f16(
                false, af.v, false, bf[j].v, (short)0, acc[j], false, false);
        }
    }
#pragma unroll
    for (int j = 0; j < 4; ++j) {
        const int n = tn + j * 16 + l15;
        const float bv = bias ? bias[n] : 0.0f;
#pragma unroll
        for (int i = 0; i < 8; ++i) {
            const int m = tm + (hi << 3) + i;
            float v = acc[j][i] + bv;
            if (relu) v = fmaxf(v, 0.0f);
            if (outF) outF[(size_t)m * Nc + n] = v;
            else      outH[(size_t)m * Nc + n] = (h16)v;
        }
    }
}

// ---------------------------------------------------------------------------
// Farthest point sampling: one block per batch, LDS distance array,
// 511 sequential argmax rounds (index 0 seeds, matching the reference scan).
// ---------------------------------------------------------------------------
__global__ void fps_kernel(const float* __restrict__ xyz, int* __restrict__ fps_idx)
{
    const int b = blockIdx.x, tid = threadIdx.x;
    __shared__ float dist[NPTS];
    __shared__ float rv[256];
    __shared__ int   ri[256];
    __shared__ float cxyz[3];
    for (int i = tid; i < NPTS; i += 256) dist[i] = 1e10f;
    if (tid == 0) fps_idx[b * NGRP] = 0;
    int cur = 0;
    __syncthreads();
    for (int s = 1; s < NGRP; ++s) {
        if (tid == 0) {
            cxyz[0] = xyz[(size_t)(b * NPTS + cur) * 3 + 0];
            cxyz[1] = xyz[(size_t)(b * NPTS + cur) * 3 + 1];
            cxyz[2] = xyz[(size_t)(b * NPTS + cur) * 3 + 2];
        }
        __syncthreads();
        float bm = -1.0f; int ba = 0;
        for (int i = tid; i < NPTS; i += 256) {
            float dx = xyz[(size_t)(b * NPTS + i) * 3 + 0] - cxyz[0];
            float dy = xyz[(size_t)(b * NPTS + i) * 3 + 1] - cxyz[1];
            float dz = xyz[(size_t)(b * NPTS + i) * 3 + 2] - cxyz[2];
            float nd = fminf(dist[i], dx * dx + dy * dy + dz * dz);
            dist[i] = nd;
            if (nd > bm) { bm = nd; ba = i; }
        }
        rv[tid] = bm; ri[tid] = ba;
        __syncthreads();
        for (int off = 128; off > 0; off >>= 1) {
            if (tid < off && rv[tid + off] > rv[tid]) { rv[tid] = rv[tid + off]; ri[tid] = ri[tid + off]; }
            __syncthreads();
        }
        cur = ri[0];
        if (tid == 0) fps_idx[b * NGRP + s] = cur;
        __syncthreads();
    }
}

// Gather centroid xyz / features; also writes new_xyz to d_out.
__global__ void gather_centers(const float* __restrict__ xyz, const float* __restrict__ pts,
                               const int* __restrict__ fps_idx,
                               float* __restrict__ nxyz, float* __restrict__ npts0,
                               float* __restrict__ out_xyz)
{
    const int t = blockIdx.x * blockDim.x + threadIdx.x;
    if (t >= BS) return;
    const int b = t / NGRP;
    const int id = fps_idx[t];
#pragma unroll
    for (int c = 0; c < 3; ++c) {
        const float v = xyz[(size_t)(b * NPTS + id) * 3 + c];
        nxyz[(size_t)t * 3 + c] = v;
        out_xyz[(size_t)t * 3 + c] = v;
    }
    for (int c = 0; c < CH; ++c)
        npts0[(size_t)t * CH + c] = pts[(size_t)(b * NPTS + id) * CH + c];
}

// kNN: one block per query; LDS distances + K iterative argmin rounds.
// (Softmax/sum over K are permutation-invariant, so selection order is free.)
__global__ void knn_kernel(const float* __restrict__ xyz, const float* __restrict__ nxyz,
                           int* __restrict__ gidx, int K)
{
    const int q = blockIdx.x;
    const int b = q / NGRP;
    const int tid = threadIdx.x;
    __shared__ float dist[NPTS];
    __shared__ float rv[256];
    __shared__ int   ri[256];
    const float qx = nxyz[(size_t)q * 3 + 0];
    const float qy = nxyz[(size_t)q * 3 + 1];
    const float qz = nxyz[(size_t)q * 3 + 2];
    for (int i = tid; i < NPTS; i += 256) {
        float dx = xyz[(size_t)(b * NPTS + i) * 3 + 0] - qx;
        float dy = xyz[(size_t)(b * NPTS + i) * 3 + 1] - qy;
        float dz = xyz[(size_t)(b * NPTS + i) * 3 + 2] - qz;
        dist[i] = dx * dx + dy * dy + dz * dz;
    }
    __syncthreads();
    for (int k = 0; k < K; ++k) {
        float bm = 3.0e38f; int ba = 0;
        for (int i = tid; i < NPTS; i += 256) {
            float d = dist[i];
            if (d < bm) { bm = d; ba = i; }
        }
        rv[tid] = bm; ri[tid] = ba;
        __syncthreads();
        for (int off = 128; off > 0; off >>= 1) {
            if (tid < off && rv[tid + off] < rv[tid]) { rv[tid] = rv[tid + off]; ri[tid] = ri[tid + off]; }
            __syncthreads();
        }
        if (tid == 0) { gidx[(size_t)q * K + k] = ri[0]; dist[ri[0]] = 3.0e38f; }
        __syncthreads();
    }
}

// Per-batch sum/sumsq of (gp - anchor) for std (ddof=1) — block reduce + atomics.
__global__ void anchor_stats(const float* __restrict__ xyz, const float* __restrict__ pts,
                             const float* __restrict__ nxyz, const float* __restrict__ npts0,
                             const int* __restrict__ gidx, float* __restrict__ bstat, int K)
{
    const int b = blockIdx.x >> 6;
    const int blk = blockIdx.x & 63;
    const int tid = threadIdx.x;
    const int E = NGRP * K * 67;
    const int per = (E + 63) / 64;
    const int e0 = blk * per;
    int e1 = e0 + per; if (e1 > E) e1 = E;
    float s = 0.f, q = 0.f;
    for (int e = e0 + tid; e < e1; e += 256) {
        const int c  = e % 67;
        const int sk = e / 67;
        const int k  = sk % K;
        const int ss = sk / K;
        const int row = b * NGRP + ss;
        const int id  = gidx[(size_t)row * K + k];
        float dif;
        if (c < CH) dif = pts[(size_t)(b * NPTS + id) * CH + c] - npts0[(size_t)row * CH + c];
        else        dif = xyz[(size_t)(b * NPTS + id) * 3 + (c - CH)] - nxyz[(size_t)row * 3 + (c - CH)];
        s += dif; q += dif * dif;
    }
    __shared__ float rs[256], rq[256];
    rs[tid] = s; rq[tid] = q;
    __syncthreads();
    for (int off = 128; off > 0; off >>= 1) {
        if (tid < off) { rs[tid] += rs[tid + off]; rq[tid] += rq[tid + off]; }
        __syncthreads();
    }
    if (tid == 0) { atomicAdd(&bstat[b * 2], rs[0]); atomicAdd(&bstat[b * 2 + 1], rq[0]); }
}

// Build normalized group features gp[M, ldg] (f16, zero-padded K for WMMA) + rel.
__global__ void build_gp(const float* __restrict__ xyz, const float* __restrict__ pts,
                         const float* __restrict__ nxyz, const float* __restrict__ npts0,
                         const int* __restrict__ gidx,
                         const float* __restrict__ alpha, const float* __restrict__ beta,
                         const float* __restrict__ bstat,
                         h16* __restrict__ XA, float* __restrict__ relbuf, int K, int ldg)
{
    const int r = blockIdx.x * blockDim.x + threadIdx.x;
    if (r >= B_ * NGRP * K) return;
    const int row = r / K;
    const int b = row / NGRP;
    const int id = gidx[r];
    const float E = (float)(NGRP * K * 67);
    const float sm = bstat[b * 2], sq = bstat[b * 2 + 1];
    const float mean = sm / E;
    const float var = (sq - sm * mean) / (E - 1.0f);
    const float inv = 1.0f / (sqrtf(var) + 1e-5f);
    float rel[3];
#pragma unroll
    for (int j = 0; j < 3; ++j) {
        rel[j] = xyz[(size_t)(b * NPTS + id) * 3 + j] - nxyz[(size_t)row * 3 + j];
        relbuf[(size_t)r * 3 + j] = rel[j];
    }
    h16* o = XA + (size_t)r * ldg;
    for (int c = 0; c < CH; ++c) {
        float dif = pts[(size_t)(b * NPTS + id) * CH + c] - npts0[(size_t)row * CH + c];
        o[c] = (h16)(alpha[c] * (dif * inv) + beta[c]);
    }
#pragma unroll
    for (int j = 0; j < 3; ++j) o[CH + j] = (h16)(alpha[CH + j] * (rel[j] * inv) + beta[CH + j]);
#pragma unroll
    for (int j = 0; j < 3; ++j) o[67 + j] = (h16)rel[j];
    for (int c = 70; c < ldg; ++c) o[c] = (h16)0.0f;
}

// pos-MLP layer 1: relu(rel @ dw1^T + db1) — only 3 MACs/out, elementwise kernel.
__global__ void pos_l1(const float* __restrict__ relbuf, const float* __restrict__ dw1,
                       const float* __restrict__ db1, h16* __restrict__ PH1, int d, int total)
{
    const int t = blockIdx.x * blockDim.x + threadIdx.x;
    if (t >= total) return;
    const int r = t / d, j = t % d;
    const float v = db1[j] + dw1[j * 3 + 0] * relbuf[(size_t)r * 3 + 0]
                          + dw1[j * 3 + 1] * relbuf[(size_t)r * 3 + 1]
                          + dw1[j * 3 + 2] * relbuf[(size_t)r * 3 + 2];
    PH1[t] = (h16)fmaxf(v, 0.0f);
}

// Training-mode BN stats (per channel over all M rows). C in {64,128,256}.
__global__ void bn_stats(const float* __restrict__ Y, float* __restrict__ st, int M, int C)
{
    const int gid = blockIdx.x * blockDim.x + threadIdx.x;
    const int c = gid % C;
    int r = gid / C;
    const int rstep = (gridDim.x * blockDim.x) / C;
    float s = 0.f, q = 0.f;
    for (; r < M; r += rstep) { float v = Y[(size_t)r * C + c]; s += v; q += v * v; }
    atomicAdd(&st[c * 2], s);
    atomicAdd(&st[c * 2 + 1], q);
}

__global__ void bn_apply(const float* __restrict__ Y, const float* __restrict__ st,
                         const float* __restrict__ g, const float* __restrict__ bb,
                         h16* __restrict__ out, int M, int C, int total)
{
    const int t = blockIdx.x * blockDim.x + threadIdx.x;
    if (t >= total) return;
    const int c = t % C;
    const float mean = st[c * 2] / (float)M;
    const float var  = st[c * 2 + 1] / (float)M - mean * mean;
    const float v = (Y[t] - mean) * rsqrtf(var + 1e-5f) * g[c] + bb[c];
    out[t] = (h16)fmaxf(v, 0.0f);
}

// t = q - k + pos (f16, attention-MLP input); vp = v + pos (in place).
__global__ void make_t_vp(const float* __restrict__ Qf, const float* __restrict__ Kf,
                          const float* __restrict__ Pf, float* __restrict__ Vf,
                          h16* __restrict__ T, int K, int d, int total)
{
    const int t = blockIdx.x * blockDim.x + threadIdx.x;
    if (t >= total) return;
    const int r = t / d, j = t % d;
    const int bs = r / K;
    T[t] = (h16)(Qf[(size_t)bs * d + j] - Kf[t] + Pf[t]);
    Vf[t] += Pf[t];
}

// softmax over K (scaled by 1/sqrt(d)) + weighted sum of vp -> cat slice.
__global__ void attn_out(const float* __restrict__ Abuf, const float* __restrict__ VP,
                         float* __restrict__ cat, int K, int d, int off)
{
    const int bs = blockIdx.x;
    const int f  = threadIdx.x;
    if (f >= d) return;
    const float sc = rsqrtf((float)d);
    const size_t base = (size_t)bs * K;
    float mx = -3.0e38f;
    for (int k = 0; k < K; ++k) mx = fmaxf(mx, Abuf[(base + k) * d + f] * sc);
    float den = 0.f, num = 0.f;
    for (int k = 0; k < K; ++k) {
        const float e = __expf(Abuf[(base + k) * d + f] * sc - mx);
        den += e;
        num += e * VP[(base + k) * d + f];
    }
    cat[(size_t)bs * CATD + off + f] = num / den;
}

// Broadcast cat [BS,384] -> out [B,S,32,384].
__global__ void bcast(const float* __restrict__ cat, float* __restrict__ out, long total)
{
    const long t = (long)blockIdx.x * blockDim.x + threadIdx.x;
    if (t >= total) return;
    const int f = (int)(t % CATD);
    const long bs = t / ((long)CATD * KOUT);
    out[t] = cat[bs * CATD + f];
}

__global__ void zero_f32(float* __restrict__ p, int n)
{
    const int t = blockIdx.x * blockDim.x + threadIdx.x;
    if (t < n) p[t] = 0.0f;
}

// f32 -> f16 with row padding (weights + activations).
__global__ void cvt_f16(const float* __restrict__ src, h16* __restrict__ dst,
                        int rows, int cols, int ldp)
{
    const int t = blockIdx.x * blockDim.x + threadIdx.x;
    if (t >= rows * ldp) return;
    const int r = t / ldp, c = t % ldp;
    dst[t] = (c < cols) ? (h16)src[(size_t)r * cols + c] : (h16)0.0f;
}

// ---------------------------------------------------------------------------
// Host-side orchestration
// ---------------------------------------------------------------------------
extern "C" void kernel_launch(void* const* d_in, const int* in_sizes, int n_in,
                              void* d_out, int out_size, void* d_ws, size_t ws_size,
                              hipStream_t stream)
{
    (void)in_sizes; (void)n_in; (void)out_size; (void)ws_size;
    const float* xyz   = (const float*)d_in[0];
    const float* pts   = (const float*)d_in[1];
    const float* alpha = (const float*)d_in[2];
    const float* beta  = (const float*)d_in[3];
    auto F = [&](int i) { return (const float*)d_in[i]; };

    // params pytree leaf order (dict keys sorted): per branch 21 leaves:
    // convs[0]{b,bb,g,w}, convs[1]{b,bb,g,w}, db1, db2, dw1, dw2, fc1_b, fc1_w,
    // gb1, gb2, gw1, gw2, wk, wq, wv
    struct BP {
        const float *c1b, *c1bb, *c1g, *c1w, *c2b, *c2bb, *c2g, *c2w;
        const float *db1, *db2, *dw1, *dw2, *fc1b, *fc1w;
        const float *gb1, *gb2, *gw1, *gw2, *wk, *wq, *wv;
    } bp[2];
    for (int br = 0; br < 2; ++br) {
        const int o = 4 + br * 21;
        bp[br] = { F(o+0), F(o+1), F(o+2), F(o+3), F(o+4), F(o+5), F(o+6), F(o+7),
                   F(o+8), F(o+9), F(o+10), F(o+11), F(o+12), F(o+13),
                   F(o+14), F(o+15), F(o+16), F(o+17), F(o+18), F(o+19), F(o+20) };
    }

    const int Ks[2]  = {16, 32};
    const int ds[2]  = {128, 256};
    const int c1o[2] = {64, 128};
    const int Mmax = BS * KOUT;      // 131072
    const int Dmax = 256;

    // workspace bump allocator
    char* wsp = (char*)d_ws;
    auto alloc = [&](size_t bytes) -> char* {
        char* r = wsp; wsp += (bytes + 255) & ~(size_t)255; return r;
    };
    float* nxyz   = (float*)alloc((size_t)BS * 3 * 4);
    float* npts0  = (float*)alloc((size_t)BS * CH * 4);
    h16*   npts0h = (h16*)  alloc((size_t)BS * CH * 2);
    int*   fpsidx = (int*)  alloc((size_t)BS * 4);
    int*   gidx   = (int*)  alloc((size_t)Mmax * 4);
    float* bstat  = (float*)alloc(16 * 4);
    float* bnstat = (float*)alloc((size_t)2 * Dmax * 4);
    float* relbuf = (float*)alloc((size_t)Mmax * 3 * 4);
    h16*   QH1    = (h16*)  alloc((size_t)BS * Dmax * 2);
    float* Qf     = (float*)alloc((size_t)BS * Dmax * 4);
    h16*   XA     = (h16*)  alloc((size_t)Mmax * Dmax * 2);
    h16*   XB     = (h16*)  alloc((size_t)Mmax * Dmax * 2);
    h16*   PH1    = (h16*)  alloc((size_t)Mmax * Dmax * 2);
    float* Y      = (float*)alloc((size_t)Mmax * Dmax * 4);   // conv out / k / attn
    float* Vf     = (float*)alloc((size_t)Mmax * Dmax * 4);   // v -> v+pos
    float* Pf     = (float*)alloc((size_t)Mmax * Dmax * 4);   // pos
    float* catb   = (float*)alloc((size_t)BS * CATD * 4);
    // f16 weight pool
    h16* w_c1  = (h16*)alloc((size_t)Dmax * 96 * 2);
    h16* w_c2  = (h16*)alloc((size_t)Dmax * Dmax * 2);
    h16* w_fc1 = (h16*)alloc((size_t)Dmax * CH * 2);
    h16* w_wq  = (h16*)alloc((size_t)Dmax * Dmax * 2);
    h16* w_wk  = (h16*)alloc((size_t)Dmax * Dmax * 2);
    h16* w_wv  = (h16*)alloc((size_t)Dmax * Dmax * 2);
    h16* w_dw2 = (h16*)alloc((size_t)Dmax * Dmax * 2);
    h16* w_gw1 = (h16*)alloc((size_t)Dmax * Dmax * 2);
    h16* w_gw2 = (h16*)alloc((size_t)Dmax * Dmax * 2);

    float* out_xyz = (float*)d_out;                  // [B,S,3]
    float* out_np  = (float*)d_out + (size_t)BS * 3; // [B,S,32,384]

    auto gemm = [&](const h16* A, const h16* W, const float* bias,
                    int M, int Nc, int Kp, float* oF, h16* oH, int relu) {
        dim3 g(M / 64, Nc / 64);
        const size_t shbytes = (size_t)Nc * Kp * sizeof(h16);   // whole W in LDS
        gemm_wmma<<<g, 128, shbytes, stream>>>(A, W, bias, oF, oH, M, Nc, Kp, relu);
    };
    auto cvt = [&](const float* s, h16* dst, int rows, int cols, int ldp) {
        cvt_f16<<<cdiv(rows * ldp, 256), 256, 0, stream>>>(s, dst, rows, cols, ldp);
    };

    // --- sampling + centroid gather (shared by both branches) ---
    fps_kernel<<<B_, 256, 0, stream>>>(xyz, fpsidx);
    gather_centers<<<cdiv(BS, 256), 256, 0, stream>>>(xyz, pts, fpsidx, nxyz, npts0, out_xyz);
    cvt(npts0, npts0h, BS, CH, CH);

    int off = 0;
    for (int br = 0; br < 2; ++br) {
        const BP& P = bp[br];
        const int K = Ks[br], d = ds[br], co1 = c1o[br];
        const int M = BS * K;

        knn_kernel<<<BS, 256, 0, stream>>>(xyz, nxyz, gidx, K);

        zero_f32<<<1, 32, 0, stream>>>(bstat, 16);
        anchor_stats<<<B_ * 64, 256, 0, stream>>>(xyz, pts, nxyz, npts0, gidx, bstat, K);
        build_gp<<<cdiv(M, 256), 256, 0, stream>>>(xyz, pts, nxyz, npts0, gidx,
                                                   alpha, beta, bstat, XA, relbuf, K, 96);

        // weight conversion (f32 -> f16, K-padded)
        cvt(P.c1w, w_c1, co1, 70, 96);
        cvt(P.c2w, w_c2, d, co1, co1);
        cvt(P.fc1w, w_fc1, d, CH, CH);
        cvt(P.wq, w_wq, d, d, d);
        cvt(P.wk, w_wk, d, d, d);
        cvt(P.wv, w_wv, d, d, d);
        cvt(P.dw2, w_dw2, d, d, d);
        cvt(P.gw1, w_gw1, d, d, d);
        cvt(P.gw2, w_gw2, d, d, d);

        // q = wq(fc1(new_points0))
        gemm(npts0h, w_fc1, P.fc1b, BS, d, CH, nullptr, QH1, 0);
        gemm(QH1, w_wq, nullptr, BS, d, d, Qf, nullptr, 0);

        // conv1 + BN + relu
        gemm(XA, w_c1, P.c1b, M, co1, 96, Y, nullptr, 0);
        zero_f32<<<cdiv(2 * co1, 256), 256, 0, stream>>>(bnstat, 2 * co1);
        bn_stats<<<256, 256, 0, stream>>>(Y, bnstat, M, co1);
        bn_apply<<<cdiv(M * co1, 256), 256, 0, stream>>>(Y, bnstat, P.c1g, P.c1bb, XB, M, co1, M * co1);

        // conv2 + BN + relu -> gp2 (f16)
        gemm(XB, w_c2, P.c2b, M, d, co1, Y, nullptr, 0);
        zero_f32<<<cdiv(2 * d, 256), 256, 0, stream>>>(bnstat, 2 * d);
        bn_stats<<<256, 256, 0, stream>>>(Y, bnstat, M, d);
        bn_apply<<<cdiv(M * d, 256), 256, 0, stream>>>(Y, bnstat, P.c2g, P.c2bb, XA, M, d, M * d);

        // pos MLP
        pos_l1<<<cdiv(M * d, 256), 256, 0, stream>>>(relbuf, P.dw1, P.db1, PH1, d, M * d);
        gemm(PH1, w_dw2, P.db2, M, d, d, Pf, nullptr, 0);

        // k, v projections of gp2
        gemm(XA, w_wk, nullptr, M, d, d, Y, nullptr, 0);   // k -> Y
        gemm(XA, w_wv, nullptr, M, d, d, Vf, nullptr, 0);  // v -> Vf

        // t = q - k + pos ; vp = v + pos
        make_t_vp<<<cdiv(M * d, 256), 256, 0, stream>>>(Qf, Y, Pf, Vf, XB, K, d, M * d);

        // attention MLP
        gemm(XB, w_gw1, P.gb1, M, d, d, nullptr, XA, 1);   // relu hidden (f16)
        gemm(XA, w_gw2, P.gb2, M, d, d, Y, nullptr, 0);    // logits -> Y

        attn_out<<<BS, d, 0, stream>>>(Y, Vf, catb, K, d, off);
        off += d;
    }

    const long total = (long)BS * KOUT * CATD;
    bcast<<<cdiv((unsigned)total, 256), 256, 0, stream>>>(catb, out_np, total);
}
